// PointNet_SA_Module_KNN_49572512530938
// MI455X (gfx1250) — matrix operand order repaired and checked by reading, not compiled, not run on hardware
//
#include <hip/hip_runtime.h>
#include <hip/hip_bf16.h>

typedef __attribute__((ext_vector_type(16))) _Float16 v16h;
typedef __attribute__((ext_vector_type(8)))  _Float16 v8h;
typedef __attribute__((ext_vector_type(2)))  _Float16 h2;
typedef __attribute__((ext_vector_type(8)))  float    v8f;

constexpr int B_ = 8, N_ = 2048, NPOINT_ = 512, K_ = 16, CIN_ = 64;
constexpr int PK_ = NPOINT_ * K_;   // 8192
constexpr int J_  = B_ * PK_;       // 65536

// ---- workspace layout (float units) ----
constexpr size_t OFF_FIDX  = 0;                       // int  B*NPOINT        = 4096
constexpr size_t OFF_KNN   = 4096;                    // int  B*NPOINT*K      = 65536
constexpr size_t OFF_GEOM  = OFF_KNN  + 65536;        // f32  B*10*PK         = 655360
constexpr size_t OFF_S     = OFF_GEOM + 655360;       // f32  B*8*PK          = 524288
constexpr size_t OFF_STATS = OFF_S    + 524288;       // f32  2*256           = 512
constexpr size_t OFF_WH    = OFF_STATS+ 512;          // f16  up to 256*1024  (=131072 f32 slots)
constexpr size_t OFF_FEATS = OFF_WH   + 131072;       // f32  up to B*128*PK  = 8388608
constexpr size_t OFF_OUT   = OFF_FEATS+ 8388608;      // f32  up to B*256*PK  = 16777216
// knn chunk scratch overlaps OFF_FEATS (dead until gather_points)

// ---------------- FPS: wave32 shuffle argmax, 2 barriers/iter ----------------
__global__ void fps_kernel(const float* __restrict__ xyz, int* __restrict__ fidx) {
  const int b = blockIdx.x, tid = threadIdx.x; // 512 threads = 16 waves
  const int lane = tid & 31, wv = tid >> 5;
  const float* xb = xyz + (size_t)b * 3 * N_;
  float px[4], py[4], pz[4], dist[4];
#pragma unroll
  for (int q = 0; q < 4; ++q) {
    int n = tid + q * 512;
    px[q] = xb[n]; py[q] = xb[N_ + n]; pz[q] = xb[2 * N_ + n];
    dist[q] = 1e10f;
  }
  __shared__ float wval[16];
  __shared__ int   widx[16];
  __shared__ int   sfar;
  int far = 0;
  for (int i = 0; i < NPOINT_; ++i) {
    if (tid == 0) fidx[b * NPOINT_ + i] = far;
    float cx = xb[far], cy = xb[N_ + far], cz = xb[2 * N_ + far];
    float bestv = -1.0f; int besti = 0;
#pragma unroll
    for (int q = 0; q < 4; ++q) {
      float dx = px[q] - cx, dy = py[q] - cy, dz = pz[q] - cz;
      float d = dx * dx + dy * dy + dz * dz;
      dist[q] = fminf(dist[q], d);
      if (dist[q] > bestv) { bestv = dist[q]; besti = tid + q * 512; }
    }
    // wave32 argmax (tie -> smaller index)
#pragma unroll
    for (int off = 16; off > 0; off >>= 1) {
      float v2 = __shfl_xor(bestv, off, 32);
      int   i2 = __shfl_xor(besti, off, 32);
      if (v2 > bestv || (v2 == bestv && i2 < besti)) { bestv = v2; besti = i2; }
    }
    if (lane == 0) { wval[wv] = bestv; widx[wv] = besti; }
    __syncthreads();
    if (wv == 0) {
      float v = (lane < 16) ? wval[lane] : -1.0f;
      int  ix = (lane < 16) ? widx[lane] : 0;
#pragma unroll
      for (int off = 8; off > 0; off >>= 1) {
        float v2 = __shfl_xor(v, off, 32);
        int   i2 = __shfl_xor(ix, off, 32);
        if (v2 > v || (v2 == v && i2 < ix)) { v = v2; ix = i2; }
      }
      if (lane == 0) sfar = ix;
    }
    __syncthreads();
    far = sfar;
  }
}

// ---------------- KNN part: top-16 over a 512-candidate chunk ----------------
__global__ void knn_part_kernel(const float* __restrict__ xyz, const int* __restrict__ fidx,
                                float* __restrict__ kd, int* __restrict__ ki) {
  int t = blockIdx.x * blockDim.x + threadIdx.x;
  if (t >= B_ * NPOINT_ * 4) return;
  int chunk = t & 3, qp = t >> 2;
  int b = qp / NPOINT_;
  const float* xb = xyz + (size_t)b * 3 * N_;
  int fi = fidx[qp];
  float ax = xb[fi], ay = xb[N_ + fi], az = xb[2 * N_ + fi];
  float a2 = ax * ax + ay * ay + az * az;
  float bd[16]; int bi[16];
#pragma unroll
  for (int q = 0; q < 16; ++q) { bd[q] = 3.4e38f; bi[q] = 0; }
  int n0 = chunk * 512;
  for (int n = n0; n < n0 + 512; ++n) {
    float x = xb[n], y = xb[N_ + n], z = xb[2 * N_ + n];
    float d = a2 + x * x + y * y + z * z - 2.0f * (ax * x + ay * y + az * z);
    if (d < bd[15]) {
      bd[15] = d; bi[15] = n;
#pragma unroll
      for (int q = 15; q > 0; --q) {
        if (bd[q] < bd[q - 1]) {
          float tv = bd[q]; bd[q] = bd[q - 1]; bd[q - 1] = tv;
          int ti = bi[q]; bi[q] = bi[q - 1]; bi[q - 1] = ti;
        }
      }
    }
  }
  size_t base = (size_t)(qp * 4 + chunk) * 16;
#pragma unroll
  for (int q = 0; q < 16; ++q) { kd[base + q] = bd[q]; ki[base + q] = bi[q]; }
}

// ---------------- KNN merge: 4 sorted lists of 16 -> top-16 ----------------
__global__ void knn_merge_kernel(const float* __restrict__ kd, const int* __restrict__ ki,
                                 int* __restrict__ knn) {
  int t = blockIdx.x * blockDim.x + threadIdx.x;
  if (t >= B_ * NPOINT_) return;
  const float* d0 = kd + (size_t)t * 64;
  const int*   i0 = ki + (size_t)t * 64;
  int pos[4] = {0, 0, 0, 0};
  for (int q = 0; q < 16; ++q) {
    float bv = 3.5e38f; int bc = 0;
#pragma unroll
    for (int c = 0; c < 4; ++c) {
      float v = d0[c * 16 + pos[c]];
      if (v < bv) { bv = v; bc = c; }
    }
    knn[t * 16 + q] = i0[bc * 16 + pos[bc]];
    pos[bc]++;
  }
}

// ---------------- geometry features + small outputs ----------------
__global__ void geom_kernel(const float* __restrict__ xyz, const float* __restrict__ nrm,
                            const float* __restrict__ Xax, const float* __restrict__ Yax,
                            const int* __restrict__ fidx, const int* __restrict__ knn,
                            float* __restrict__ geom, float* __restrict__ o_xyz,
                            float* __restrict__ o_nrm, float* __restrict__ o_X, float* __restrict__ o_Y) {
  int t = blockIdx.x * blockDim.x + threadIdx.x;
  if (t >= B_ * NPOINT_) return;
  int b = t / NPOINT_, p = t % NPOINT_;
  const float* xb = xyz + (size_t)b * 3 * N_;
  const float* nb = nrm + (size_t)b * 3 * N_;
  const float* Xb = Xax + (size_t)b * 3 * N_;
  const float* Yb = Yax + (size_t)b * 3 * N_;
  int fi = fidx[t];
  float cx = xb[fi], cy = xb[N_ + fi], cz = xb[2 * N_ + fi];
  o_xyz[(b * 3 + 0) * NPOINT_ + p] = cx;
  o_xyz[(b * 3 + 1) * NPOINT_ + p] = cy;
  o_xyz[(b * 3 + 2) * NPOINT_ + p] = cz;

  float gx[16], gy[16], gz[16], dd[16];
  float dmin = 3.4e38f, dmax = -3.4e38f;
#pragma unroll
  for (int kk = 0; kk < 16; ++kk) {
    int i = knn[t * 16 + kk];
    gx[kk] = xb[i] - cx; gy[kk] = xb[N_ + i] - cy; gz[kk] = xb[2 * N_ + i] - cz;
    float d = sqrtf(gx[kk] * gx[kk] + gy[kk] * gy[kk] + gz[kk] * gz[kk] + 1e-10f);
    dd[kk] = d; dmin = fminf(dmin, d); dmax = fmaxf(dmax, d);
  }
  float invrng = 1.0f / (dmax - dmin + 1e-10f);
  const float INV_PI = 0.31830988618379067f;
  size_t gb = ((size_t)b * 10) * PK_ + (size_t)p * 16;
#pragma unroll
  for (int kk = 0; kk < 16; ++kk) {
    int i = knn[t * 16 + kk];
    float nx = nb[i], ny = nb[N_ + i], nz = nb[2 * N_ + i];
    float xx = Xb[i], xy = Xb[N_ + i], xz = Xb[2 * N_ + i];
    float yx = Yb[i], yy = Yb[N_ + i], yz = Yb[2 * N_ + i];
    float nn = sqrtf(gx[kk] * gx[kk] + gy[kk] * gy[kk] + gz[kk] * gz[kk]);
    float cn = (gx[kk] * nx + gy[kk] * ny + gz[kk] * nz) /
               (nn * sqrtf(nx * nx + ny * ny + nz * nz) + 1e-8f);
    float cX = (gx[kk] * xx + gy[kk] * xy + gz[kk] * xz) /
               (nn * sqrtf(xx * xx + xy * xy + xz * xz) + 1e-8f);
    float cY = (gx[kk] * yx + gy[kk] * yy + gz[kk] * yz) /
               (nn * sqrtf(yx * yx + yy * yy + yz * yz) + 1e-8f);
    cn = fminf(fmaxf(cn, -1.0f), 1.0f);
    cX = fminf(fmaxf(cX, -1.0f), 1.0f);
    cY = fminf(fmaxf(cY, -1.0f), 1.0f);
    geom[gb + 0 * PK_ + kk] = gx[kk];
    geom[gb + 1 * PK_ + kk] = gy[kk];
    geom[gb + 2 * PK_ + kk] = gz[kk];
    geom[gb + 3 * PK_ + kk] = (dd[kk] - dmin) * invrng;
    geom[gb + 4 * PK_ + kk] = nx;
    geom[gb + 5 * PK_ + kk] = ny;
    geom[gb + 6 * PK_ + kk] = nz;
    geom[gb + 7 * PK_ + kk] = acosf(cn) * INV_PI;
    geom[gb + 8 * PK_ + kk] = acosf(cX) * INV_PI;
    geom[gb + 9 * PK_ + kk] = acosf(cY) * INV_PI;
    if (kk == 0) {
      o_nrm[(b * 3 + 0) * NPOINT_ + p] = nx; o_nrm[(b * 3 + 1) * NPOINT_ + p] = ny; o_nrm[(b * 3 + 2) * NPOINT_ + p] = nz;
      o_X[(b * 3 + 0) * NPOINT_ + p] = xx;  o_X[(b * 3 + 1) * NPOINT_ + p] = xy;  o_X[(b * 3 + 2) * NPOINT_ + p] = xz;
      o_Y[(b * 3 + 0) * NPOINT_ + p] = yx;  o_Y[(b * 3 + 1) * NPOINT_ + p] = yy;  o_Y[(b * 3 + 2) * NPOINT_ + p] = yz;
    }
  }
}

// ---------------- gather point features ----------------
__global__ void gather_points_kernel(const float* __restrict__ points, const int* __restrict__ knn,
                                     float* __restrict__ feats) {
  int t = blockIdx.x * blockDim.x + threadIdx.x;
  if (t >= B_ * CIN_ * NPOINT_) return;
  int b = t / (CIN_ * NPOINT_), c = (t / NPOINT_) % CIN_, p = t % NPOINT_;
  const int* kn = knn + ((size_t)b * NPOINT_ + p) * 16;
  const float* src = points + ((size_t)b * CIN_ + c) * N_;
  float* dst = feats + ((size_t)b * CIN_ + c) * PK_ + (size_t)p * 16;
#pragma unroll
  for (int kk = 0; kk < 16; ++kk) dst[kk] = src[kn[kk]];
}

// ---------------- scores: h = relu(w1 g + b1), s = softmax(w2 h + b2) ----------------
__global__ void score_kernel(const float* __restrict__ geom,
                             const float* __restrict__ w1, const float* __restrict__ b1,
                             const float* __restrict__ w2, const float* __restrict__ b2,
                             float* __restrict__ sbuf) {
  int t = blockIdx.x * blockDim.x + threadIdx.x;
  if (t >= J_) return;
  int b = t / PK_, r = t % PK_;
  float g[10];
#pragma unroll
  for (int q = 0; q < 10; ++q) g[q] = geom[((size_t)b * 10 + q) * PK_ + r];
  float h[16];
#pragma unroll
  for (int i = 0; i < 16; ++i) {
    float acc = b1[i];
#pragma unroll
    for (int q = 0; q < 10; ++q) acc += w1[i * 10 + q] * g[q];
    h[i] = fmaxf(acc, 0.0f);
  }
  float e[8]; float emax = -3.4e38f;
#pragma unroll
  for (int m = 0; m < 8; ++m) {
    float acc = b2[m];
#pragma unroll
    for (int i = 0; i < 16; ++i) acc += w2[m * 16 + i] * h[i];
    e[m] = acc; emax = fmaxf(emax, acc);
  }
  float esum = 0.0f;
#pragma unroll
  for (int m = 0; m < 8; ++m) { e[m] = expf(e[m] - emax); esum += e[m]; }
  float inv = 1.0f / esum;
#pragma unroll
  for (int m = 0; m < 8; ++m) sbuf[((size_t)b * 8 + m) * PK_ + r] = e[m] * inv;
}

// ---------------- pack bank (M,cin,cout) fp32 -> Wh[o][k=m*cin+c] fp16 ----------------
__global__ void pack_bank_kernel(const float* __restrict__ bank, _Float16* __restrict__ wh,
                                 int cin, int cout) {
  int t = blockIdx.x * blockDim.x + threadIdx.x;
  int KD = 8 * cin;
  if (t >= KD * cout) return;
  int o = t / KD, k = t % KD;
  wh[t] = (_Float16)bank[(size_t)k * cout + o];
}

// ---------------- fused PAConv GEMM with WMMA ----------------
// out[o,j] = sum_m s[m,j] * sum_c bank[m,c,o] feats[c,j]
// B base fragments (feats) live in registers; softmax scale applied per-m with v_pk_mul_f16.
// Scaled fragments are double-buffered per sub-tile (bs[SJT]) so WMMA WAR hazards are
// covered by useful instructions instead of v_nops.
template <int TCIN, int TCOUT, int SJT>
__global__ __launch_bounds__(TCOUT * 2) void paconv_gemm_kernel(
    const float* __restrict__ feats, const float* __restrict__ sbuf,
    const _Float16* __restrict__ wh, float* __restrict__ outb) {
  constexpr int KD = 8 * TCIN;
  constexpr int NTHR = (TCOUT / 16) * 32;
  constexpr int CB = TCIN / 32;            // 16x16x32 K-blocks per m
  constexpr int JT = SJT * 16;             // columns per block
  __shared__ _Float16 ldsF[SJT * TCIN * 16];  // pair-packed over c: [sj][(c>>1)*16+j]*2+(c&1)
  __shared__ float st[8 * JT];
  const int tid = threadIdx.x;
  const int jb = blockIdx.x;               // 0 .. J/JT-1
  const int b  = jb / (PK_ / JT);
  const int r0 = (jb % (PK_ / JT)) * JT;

  for (int e = tid; e < 8 * JT; e += NTHR)
    st[e] = sbuf[((size_t)b * 8 + (e / JT)) * PK_ + r0 + (e % JT)];
  for (int e = tid; e < TCIN * JT; e += NTHR) {
    int c = e / JT, jj = e % JT;
    int sj = jj >> 4, j = jj & 15;
    ldsF[sj * (TCIN * 16) + ((c >> 1) * 16 + j) * 2 + (c & 1)] =
        (_Float16)feats[((size_t)b * TCIN + c) * PK_ + r0 + jj];
  }
  __syncthreads();

  const int lane = tid & 31;
  const int wave = tid >> 5;
  const int obase = wave * 16;
  const int orow = obase + (lane & 15);
  const int hi = (lane >> 4) & 1;
  const int j = lane & 15;
  const unsigned* ldsW = (const unsigned*)ldsF;

  union F16x16 { v16h v; h2 p[8]; unsigned u[8]; };

  // load base B fragments once (resident across all 8 m iterations)
  F16x16 bf[CB][SJT];
#pragma unroll
  for (int cb = 0; cb < CB; ++cb)
#pragma unroll
    for (int sj = 0; sj < SJT; ++sj) {
      int kl = cb * 32 + hi * 8;
#pragma unroll
      for (int r = 0; r < 8; ++r)
        bf[cb][sj].u[r] = ldsW[sj * (TCIN * 8) + ((kl + 2 * (r & 3) + (r >> 2) * 16) >> 1) * 16 + j];
    }

  v8f acc[SJT];
#pragma unroll
  for (int sj = 0; sj < SJT; ++sj) acc[sj] = v8f{};

#pragma unroll 1
  for (int m = 0; m < 8; ++m) {
    h2 sm[SJT];
#pragma unroll
    for (int sj = 0; sj < SJT; ++sj) {
      _Float16 hv = (_Float16)st[m * JT + sj * 16 + j];
      sm[sj][0] = hv; sm[sj][1] = hv;
    }
#pragma unroll
    for (int cb = 0; cb < CB; ++cb) {
      const _Float16* pa = wh + (size_t)orow * KD + m * TCIN + cb * 32 + hi * 8;
      v8h alo = *(const v8h*)pa;
      v8h ahi = *(const v8h*)(pa + 16);
      v16h a;
#pragma unroll
      for (int i = 0; i < 8; ++i) { a[i] = alo[i]; a[i + 8] = ahi[i]; }
      // batch: compute ALL scaled fragments, then issue ALL wmmas back-to-back
      F16x16 bs[SJT];
#pragma unroll
      for (int sj = 0; sj < SJT; ++sj)
#pragma unroll
        for (int r = 0; r < 8; ++r) bs[sj].p[r] = bf[cb][sj].p[r] * sm[sj];  // v_pk_mul_f16
#pragma unroll
      for (int sj = 0; sj < SJT; ++sj)
        acc[sj] = __builtin_amdgcn_wmma_f32_16x16x32_f16(false, a, false, bs[sj].v, (short)0,
                                                         acc[sj], false, false);
    }
  }
#pragma unroll
  for (int sj = 0; sj < SJT; ++sj)
#pragma unroll
    for (int r = 0; r < 8; ++r) {
      int o = obase + r + hi * 8;
      outb[((size_t)b * TCOUT + o) * PK_ + r0 + sj * 16 + j] = acc[sj][r];
    }
}

// ---------------- batchnorm stats (per channel over b,p,k) ----------------
__global__ void bn_stats_kernel(const float* __restrict__ x, float* __restrict__ stats, int cout) {
  int o = blockIdx.x, tid = threadIdx.x;
  float s = 0.0f, ss = 0.0f;
  for (int idx = tid; idx < B_ * PK_; idx += 256) {
    int b = idx / PK_, r = idx % PK_;
    float v = x[((size_t)b * cout + o) * PK_ + r];
    s += v; ss += v * v;
  }
  __shared__ float sh1[256], sh2[256];
  sh1[tid] = s; sh2[tid] = ss;
  __syncthreads();
  for (int t = 128; t > 0; t >>= 1) {
    if (tid < t) { sh1[tid] += sh1[tid + t]; sh2[tid] += sh2[tid + t]; }
    __syncthreads();
  }
  if (tid == 0) {
    float inv = 1.0f / (float)(B_ * PK_);
    float mu = sh1[0] * inv;
    stats[2 * o] = mu;
    stats[2 * o + 1] = sh2[0] * inv - mu * mu;
  }
}

__global__ void bn_apply_kernel(const float* __restrict__ x, const float* __restrict__ stats,
                                const float* __restrict__ gamma, const float* __restrict__ beta,
                                float* __restrict__ y, int cout) {
  size_t t = (size_t)blockIdx.x * blockDim.x + threadIdx.x;
  if (t >= (size_t)B_ * cout * PK_) return;
  int o = (int)((t / PK_) % cout);
  float v = (x[t] - stats[2 * o]) * rsqrtf(stats[2 * o + 1] + 1e-5f) * gamma[o] + beta[o];
  y[t] = fmaxf(v, 0.0f);
}

// ---------------- final max over K ----------------
__global__ void final_max_kernel(const float* __restrict__ x, float* __restrict__ out) {
  int t = blockIdx.x * blockDim.x + threadIdx.x;
  if (t >= B_ * 256 * NPOINT_) return;
  const float* p = x + (size_t)t * 16;
  float m = -3.4e38f;
#pragma unroll
  for (int kk = 0; kk < 16; ++kk) m = fmaxf(m, p[kk]);
  out[t] = m;
}

extern "C" void kernel_launch(void* const* d_in, const int* in_sizes, int n_in,
                              void* d_out, int out_size, void* d_ws, size_t ws_size,
                              hipStream_t stream) {
  const float* xyz  = (const float*)d_in[0];
  const float* nrm  = (const float*)d_in[1];
  const float* Xax  = (const float*)d_in[2];
  const float* Yax  = (const float*)d_in[3];
  const float* pts  = (const float*)d_in[4];
  const float* w1[3]   = {(const float*)d_in[5],  (const float*)d_in[12], (const float*)d_in[19]};
  const float* b1[3]   = {(const float*)d_in[6],  (const float*)d_in[13], (const float*)d_in[20]};
  const float* w2[3]   = {(const float*)d_in[7],  (const float*)d_in[14], (const float*)d_in[21]};
  const float* b2[3]   = {(const float*)d_in[8],  (const float*)d_in[15], (const float*)d_in[22]};
  const float* bank[3] = {(const float*)d_in[9],  (const float*)d_in[16], (const float*)d_in[23]};
  const float* gamma[2] = {(const float*)d_in[10], (const float*)d_in[17]};
  const float* beta[2]  = {(const float*)d_in[11], (const float*)d_in[18]};

  float* ws = (float*)d_ws;
  int*      fidx  = (int*)(ws + OFF_FIDX);
  int*      knn   = (int*)(ws + OFF_KNN);
  float*    geom  = ws + OFF_GEOM;
  float*    sbuf  = ws + OFF_S;
  float*    stats = ws + OFF_STATS;
  _Float16* wh    = (_Float16*)(ws + OFF_WH);
  float*    feats = ws + OFF_FEATS;
  float*    outb  = ws + OFF_OUT;
  float*    kd    = ws + OFF_FEATS;              // overlap: dead before gather
  int*      ki    = (int*)(ws + OFF_FEATS + 262144);

  float* out = (float*)d_out;
  float* o_xyz = out;
  float* o_nrm = out + 12288;
  float* o_X   = out + 24576;
  float* o_Y   = out + 36864;
  float* o_pts = out + 49152;

  fps_kernel<<<B_, 512, 0, stream>>>(xyz, fidx);
  knn_part_kernel<<<(B_ * NPOINT_ * 4 + 255) / 256, 256, 0, stream>>>(xyz, fidx, kd, ki);
  knn_merge_kernel<<<(B_ * NPOINT_ + 255) / 256, 256, 0, stream>>>(kd, ki, knn);
  geom_kernel<<<(B_ * NPOINT_ + 255) / 256, 256, 0, stream>>>(xyz, nrm, Xax, Yax, fidx, knn,
                                                              geom, o_xyz, o_nrm, o_X, o_Y);
  gather_points_kernel<<<(B_ * CIN_ * NPOINT_ + 255) / 256, 256, 0, stream>>>(pts, knn, feats);

  const int cins[3]  = {64, 64, 128};
  const int couts[3] = {64, 128, 256};
  for (int l = 0; l < 3; ++l) {
    int cin = cins[l], cout = couts[l], KD = 8 * cin;
    score_kernel<<<J_ / 256, 256, 0, stream>>>(geom, w1[l], b1[l], w2[l], b2[l], sbuf);
    pack_bank_kernel<<<(KD * cout + 255) / 256, 256, 0, stream>>>(bank[l], wh, cin, cout);
    if (l == 0)      paconv_gemm_kernel<64, 64, 4><<<J_ / 64, 128, 0, stream>>>(feats, sbuf, wh, outb);
    else if (l == 1) paconv_gemm_kernel<64, 128, 4><<<J_ / 64, 256, 0, stream>>>(feats, sbuf, wh, outb);
    else             paconv_gemm_kernel<128, 256, 2><<<J_ / 32, 512, 0, stream>>>(feats, sbuf, wh, outb);
    if (l < 2) {
      bn_stats_kernel<<<cout, 256, 0, stream>>>(outb, stats, cout);
      size_t tot = (size_t)B_ * cout * PK_;
      bn_apply_kernel<<<(unsigned)((tot + 255) / 256), 256, 0, stream>>>(outb, stats, gamma[l],
                                                                         beta[l], feats, cout);
    }
  }
  final_max_kernel<<<(B_ * 256 * NPOINT_ + 255) / 256, 256, 0, stream>>>(outb, o_pts);
}
